// HeterGAT_41635412968140
// MI455X (gfx1250) — compile-verified, matrix operand order checked by reading, not run on hardware
//
#include <hip/hip_runtime.h>

// ---------------- problem constants (match reference) ----------------
#define N_NODES 30000
#define N_EDGES 100000
#define N_TYPES 6
#define D_IN    128
#define HC      256      // H * C
#define H       8
#define LRELU   0.2f

// ---------------- WMMA / vector types (gfx1250, wave32) ----------------
typedef __attribute__((ext_vector_type(16))) _Float16 v16h;
typedef __attribute__((ext_vector_type(8)))  _Float16 v8h;
typedef __attribute__((ext_vector_type(4)))  _Float16 v4h;
typedef __attribute__((ext_vector_type(8)))  float    v8f;
typedef __attribute__((ext_vector_type(4)))  float    v4f;
typedef __attribute__((ext_vector_type(4)))  unsigned int u32x4;
typedef __attribute__((ext_vector_type(8)))  int      i32x8;
typedef __attribute__((ext_vector_type(4)))  int      i32x4;

#if __has_builtin(__builtin_amdgcn_tensor_load_to_lds) && \
    __has_builtin(__builtin_amdgcn_s_wait_tensorcnt)
#define USE_TDM 1
#else
#define USE_TDM 0
#endif

// =====================================================================
// GEMM: C[M,256] = A[M,K] @ B[K,256]   (fp32 in/out, f16 WMMA, f32 acc)
// Block: 256 threads = 8 waves. Tile: BM=128 x BN=256, BK=32.
//  - A tile staged as raw f32 by the Tensor Data Mover (TENSOR_LOAD_TO_LDS,
//    TENSORcnt-tracked; descriptor OOB handles the M edge), converted to
//    f16 fragments in VGPRs once per K-step.
//  - B tile staged in WMMA fragment layout sB[nt][lane][j]: each lane's bf
//    is one 32B-aligned v16h -> two ds_load_b128.
// =====================================================================
#define BM 128
#define BK 32
#define BN 256

__global__ __launch_bounds__(256)
void k_gemm_wmma(const float* __restrict__ A, const float* __restrict__ B,
                 float* __restrict__ C, int M, int K) {
  __shared__ __align__(32) float    sA32[BM * BK];        // 16 KB, row-major f32
  __shared__ __align__(32) _Float16 sB[16 * 32 * 16];     // 16 KB, frag layout
  const int tid  = threadIdx.x;
  const int lane = tid & 31;
  const int wid  = tid >> 5;
  const int g    = lane >> 4;   // half-wave group (K-range selector)
  const int ml   = lane & 15;   // row (A) / col (B) within 16-tile
  const int row0 = blockIdx.x * BM;

  v8f acc[16];
#pragma unroll
  for (int i = 0; i < 16; ++i) acc[i] = (v8f)0.0f;

  for (int k0 = 0; k0 < K; k0 += BK) {
#if USE_TDM
    // --- A tile via Tensor Data Mover: 2D tile 32(f32) x 128 rows ---
    if (wid == 0) {
      unsigned lds_addr = (unsigned)(unsigned long long)(void*)&sA32[0];
      unsigned long long ga =
          (unsigned long long)(const void*)(A + (long)row0 * K + k0);
      // D# group0: count=1 | lds_addr | global_addr | type=2
      u32x4 g0;
      g0.x = 1u;                                             // count = 1
      g0.y = lds_addr;                                       // bits 63:32
      g0.z = (unsigned)(ga & 0xffffffffu);                   // ga[31:0]
      g0.w = (unsigned)((ga >> 32) & 0x01ffffffu) | 0x80000000u; // ga[56:32]|type=2
      // D# group1: data_size=4B, tensor_dim0=32, tensor_dim1=M-row0,
      // tile_dim0=32, tile_dim1=128, tensor_dim0_stride=K
      unsigned td0 = 32u;
      unsigned td1 = (unsigned)(M - row0);
      unsigned long long st0 = (unsigned long long)(unsigned)K;
      i32x8 g1;
      g1[0] = (int)(2u << 16);                               // data_size=2 (4B)
      g1[1] = (int)((td0 & 0xffffu) << 16);                  // tensor_dim0 lo
      g1[2] = (int)((td0 >> 16) | ((td1 & 0xffffu) << 16));  // td0 hi | td1 lo
      g1[3] = (int)((td1 >> 16) | (32u << 16));              // td1 hi | tile_dim0
      g1[4] = (int)128u;                                     // tile_dim1 (tile_dim2=0)
      g1[5] = (int)(st0 & 0xffffffffu);                      // dim0 stride lo
      g1[6] = (int)((st0 >> 32) & 0xffffu);                  // dim0 stride hi
      g1[7] = 0;
      i32x4 z4 = (i32x4)0;
#if __clang_major__ >= 23
      i32x8 z8 = (i32x8)0;
      __builtin_amdgcn_tensor_load_to_lds(g0, g1, z4, z4, z8, 0);
#else
      __builtin_amdgcn_tensor_load_to_lds(g0, g1, z4, z4, 0);
#endif
    }
#else
    // --- fallback: plain float4 copy of A tile ---
#pragma unroll
    for (int r = 0; r < 4; ++r) {
      int idx = tid + r * 256;            // 0..1023 float4 slots
      int ar  = idx >> 3;                 // 0..127
      int ac  = (idx & 7) * 4;            // 0,4,..,28
      int grow = row0 + ar;
      v4f v = (grow < M) ? *(const v4f*)(A + (long)grow * K + (k0 + ac))
                         : (v4f)0.0f;
      *(v4f*)&sA32[ar * BK + ac] = v;
    }
#endif
    // --- stage B tile 32x256: float4 loads, scatter to fragment layout ---
#pragma unroll
    for (int r = 0; r < 8; ++r) {
      int idx = tid + r * 256;            // 0..2047 float4 slots
      int br  = idx >> 6;                 // k within tile: 0..31
      int bc  = (idx & 63) * 4;           // n: 0,4,..,252
      v4f v = *(const v4f*)(B + (long)(k0 + br) * BN + bc);
      int nt = bc >> 4;                   // 16-col tile
      int nl = bc & 15;                   // 0,4,8,12
      int gg = (br >> 3) & 1;             // frag half-wave group for this k
      int jj = (br & 7) + ((br >= 16) ? 8 : 0);  // frag element index
      _Float16* d0 = &sB[nt * 512 + (gg * 16 + nl) * 16 + jj];
      d0[0]  = (_Float16)v.x;             // n = nl+0
      d0[16] = (_Float16)v.y;             // n = nl+1
      d0[32] = (_Float16)v.z;             // n = nl+2
      d0[48] = (_Float16)v.w;             // n = nl+3
    }
    // prefetch next K-tile B lines (global_prefetch_b8 path)
    if (k0 + BK < K) {
      __builtin_prefetch(B + (long)(k0 + BK + (tid >> 3)) * BN + (tid & 7) * 32, 0, 1);
    }
#if USE_TDM
    if (wid == 0) __builtin_amdgcn_s_wait_tensorcnt(0);
#endif
    __syncthreads();

    // A fragment: two contiguous v8f runs per lane, convert f32->f16
    const int rowbase = (wid * 16 + ml) * BK;
    v8f a_lo32 = *(const v8f*)&sA32[rowbase + g * 8];
    v8f a_hi32 = *(const v8f*)&sA32[rowbase + 16 + g * 8];
    v8h a_lo = __builtin_convertvector(a_lo32, v8h);
    v8h a_hi = __builtin_convertvector(a_hi32, v8h);
    v16h af = __builtin_shufflevector(a_lo, a_hi,
        0, 1, 2, 3, 4, 5, 6, 7, 8, 9, 10, 11, 12, 13, 14, 15);

#pragma unroll
    for (int nt = 0; nt < 16; ++nt) {
      v16h bf = *(const v16h*)&sB[nt * 512 + lane * 16];   // 32B aligned
      acc[nt] = __builtin_amdgcn_wmma_f32_16x16x32_f16(
          false, af, false, bf, (short)0, acc[nt], false, false);
    }
    __syncthreads();
  }

  // store: C/D layout — VGPR v: lanes 0-15 -> M=v, lanes 16-31 -> M=v+8
#pragma unroll
  for (int nt = 0; nt < 16; ++nt) {
    int col = nt * 16 + ml;
#pragma unroll
    for (int v = 0; v < 8; ++v) {
      int row = row0 + wid * 16 + v + 8 * g;
      if (row < M) C[(long)row * BN + col] = acc[nt][v];
    }
  }
}

// =====================================================================
// Per-node attention logits: e[n,h] = sum_c h[n, h*32+c] * att[h,c]
// =====================================================================
__global__ __launch_bounds__(256)
void k_att(const float* __restrict__ hmat, const float* __restrict__ att,
           float* __restrict__ e) {
  int i = blockIdx.x * 256 + threadIdx.x;
  if (i >= N_NODES * H) return;
  int n = i >> 3, h = i & 7;
  const float* row = hmat + (long)n * HC + h * 32;
  const float* a   = att + h * 32;
  float s = 0.f;
#pragma unroll
  for (int c = 0; c < 32; ++c) s += row[c] * a[c];
  e[i] = s;
}

// ---------------- segment softmax helpers ----------------
__device__ __forceinline__ unsigned f2key(float f) {
  unsigned u = __float_as_uint(f);
  return (u & 0x80000000u) ? ~u : (u | 0x80000000u);
}
__device__ __forceinline__ float key2f(unsigned k) {
  unsigned u = (k & 0x80000000u) ? (k ^ 0x80000000u) : ~k;
  return __uint_as_float(u);
}

__global__ __launch_bounds__(256)
void k_init_sm(unsigned* __restrict__ menc, float* __restrict__ den) {
  int i = blockIdx.x * 256 + threadIdx.x;
  if (i < N_NODES * H) { menc[i] = f2key(-__builtin_huge_valf()); den[i] = 0.f; }
}

__global__ __launch_bounds__(256)
void k_edge_max(const int* __restrict__ src, const int* __restrict__ dst,
                const float* __restrict__ es, const float* __restrict__ ed,
                unsigned* __restrict__ menc) {
  int e = blockIdx.x * 256 + threadIdx.x;
  if (e >= N_EDGES) return;
  int s = src[e], d = dst[e];
#pragma unroll
  for (int h = 0; h < H; ++h) {
    float v = es[s * H + h] + ed[d * H + h];
    v = (v > 0.f) ? v : LRELU * v;
    atomicMax(&menc[d * H + h], f2key(v));
  }
}

__global__ __launch_bounds__(256)
void k_edge_expsum(const int* __restrict__ src, const int* __restrict__ dst,
                   const float* __restrict__ es, const float* __restrict__ ed,
                   const unsigned* __restrict__ menc, float* __restrict__ den,
                   float* __restrict__ exb) {
  int e = blockIdx.x * 256 + threadIdx.x;
  if (e >= N_EDGES) return;
  int s = src[e], d = dst[e];
#pragma unroll
  for (int h = 0; h < H; ++h) {
    float v = es[s * H + h] + ed[d * H + h];
    v = (v > 0.f) ? v : LRELU * v;
    float ex = __expf(v - key2f(menc[d * H + h]));
    exb[e * H + h] = ex;
    atomicAdd(&den[d * H + h], ex);
  }
}

__global__ __launch_bounds__(256)
void k_init_out(float* __restrict__ out, const float* __restrict__ bias) {
  long i = (long)blockIdx.x * 256 + threadIdx.x;
  if (i < (long)N_NODES * HC) out[i] = bias[i & 255];
}

// one block per edge; 256 threads carry the full [H=8, C=32] message
__global__ __launch_bounds__(256)
void k_edge_agg(const int* __restrict__ src, const int* __restrict__ dst,
                const float* __restrict__ exb, const float* __restrict__ den,
                const float* __restrict__ hs, float* __restrict__ out,
                float* __restrict__ alpha_out) {
  int e = blockIdx.x;
  int tid = threadIdx.x;
  int h = tid >> 5;
  int s = src[e], d = dst[e];
  float alpha = exb[e * H + h] / (den[d * H + h] + 1e-16f);
  atomicAdd(&out[(long)d * HC + tid], alpha * hs[(long)s * HC + tid]);
  if (alpha_out != nullptr && (tid & 31) == 0) alpha_out[e * H + h] = alpha;
}

// =====================================================================
// LayerNorm(256) + ReLU, one 256-thread block per row
// =====================================================================
__global__ __launch_bounds__(256)
void k_ln_relu(float* __restrict__ y, const float* __restrict__ x,
               const float* __restrict__ g, const float* __restrict__ b) {
  __shared__ float sm[256];
  long row = blockIdx.x;
  int tid = threadIdx.x;
  float v = x[row * HC + tid];
  sm[tid] = v; __syncthreads();
  for (int st = 128; st > 0; st >>= 1) { if (tid < st) sm[tid] += sm[tid + st]; __syncthreads(); }
  float mu = sm[0] * (1.f / HC); __syncthreads();
  float xm = v - mu;
  sm[tid] = xm * xm; __syncthreads();
  for (int st = 128; st > 0; st >>= 1) { if (tid < st) sm[tid] += sm[tid + st]; __syncthreads(); }
  float var = sm[0] * (1.f / HC);
  float o = xm * rsqrtf(var + 1e-5f) * g[tid] + b[tid];
  y[row * HC + tid] = fmaxf(o, 0.f);
}

// column sums (for mean pooling): block = 32 rows, thread per column
__global__
void k_colsum(const float* __restrict__ x, float* __restrict__ out,
              int ncols, int nrows) {
  int c = threadIdx.x;
  int r0 = blockIdx.x * 32;
  float s = 0.f;
  for (int r = 0; r < 32; ++r) {
    int rr = r0 + r;
    if (rr < nrows) s += x[(long)rr * ncols + c];
  }
  atomicAdd(&out[c], s);
}

__global__ __launch_bounds__(256)
void k_zero(float* __restrict__ p, int n) {
  int i = blockIdx.x * 256 + threadIdx.x;
  if (i < n) p[i] = 0.f;
}

__global__ __launch_bounds__(256)
void k_final(const float* __restrict__ pooled, const float* __restrict__ pkg,
             const float* __restrict__ Wc, const float* __restrict__ bc,
             float* __restrict__ out0) {
  __shared__ float sm[256];
  int tid = threadIdx.x;
  const float inv = 1.0f / (float)N_NODES;
  float s = 0.f;
  for (int i = tid; i < N_TYPES * HC; i += 256) s += (pooled[i] * inv) * Wc[i];
  if (tid < D_IN) s += (pkg[tid] * inv) * Wc[N_TYPES * HC + tid];
  sm[tid] = s; __syncthreads();
  for (int st = 128; st > 0; st >>= 1) { if (tid < st) sm[tid] += sm[tid + st]; __syncthreads(); }
  if (tid == 0) out0[0] = sm[0] + bc[0];
}

// =====================================================================
// launcher
// =====================================================================
extern "C" void kernel_launch(void* const* d_in, const int* in_sizes, int n_in,
                              void* d_out, int out_size, void* d_ws, size_t ws_size,
                              hipStream_t stream) {
  (void)in_sizes; (void)n_in; (void)out_size; (void)ws_size;

  const float* x_pkg   = (const float*)d_in[0];
  const float* x_tgt   = (const float*)d_in[1];
  const int*   ei      = (const int*)  d_in[2];
  const float* W1_src  = (const float*)d_in[3];
  const float* W1_dst  = (const float*)d_in[4];
  const float* a1_src  = (const float*)d_in[5];
  const float* a1_dst  = (const float*)d_in[6];
  const float* b1      = (const float*)d_in[7];
  const float* W2_src  = (const float*)d_in[8];
  const float* W2_dst  = (const float*)d_in[9];
  const float* a2_src  = (const float*)d_in[10];
  const float* a2_dst  = (const float*)d_in[11];
  const float* b2      = (const float*)d_in[12];
  const float* gamma   = (const float*)d_in[13];
  const float* beta    = (const float*)d_in[14];
  const float* Wc      = (const float*)d_in[15];
  const float* bc      = (const float*)d_in[16];
  float* out = (float*)d_out;

  // ---- carve workspace (aligned 256B) ----
  char* w = (char*)d_ws;
  auto carve = [&](size_t bytes) -> void* {
    void* p = (void*)w;
    w += (bytes + 255) & ~((size_t)255);
    return p;
  };
  const size_t NM = (size_t)N_NODES * HC;
  float*    bufA  = (float*)   carve(NM * 4);          // hs1 / hs2
  float*    bufB  = (float*)   carve(NM * 4);          // hd1 / hd2
  float*    bufC  = (float*)   carve(NM * 4);          // out1 / out2 accum
  float*    bufD  = (float*)   carve(NM * 4);          // x1 / x2
  float*    es    = (float*)   carve((size_t)N_NODES * H * 4);
  float*    ed    = (float*)   carve((size_t)N_NODES * H * 4);
  unsigned* menc  = (unsigned*)carve((size_t)N_NODES * H * 4);
  float*    den   = (float*)   carve((size_t)N_NODES * H * 4);
  float*    exb   = (float*)   carve((size_t)N_EDGES * H * 4);
  float*    pooled= (float*)   carve((size_t)(N_TYPES * HC + D_IN) * 4);

  const int gGemm = (N_NODES + BM - 1) / BM;           // 235
  const int gNH   = (N_NODES * H + 255) / 256;         // 938
  const int gE    = (N_EDGES + 255) / 256;             // 391
  const int gRows = (N_NODES + 31) / 32;               // 938
  const int gZero = (N_TYPES * HC + D_IN + 255) / 256; // 7

  k_zero<<<gZero, 256, 0, stream>>>(pooled, N_TYPES * HC + D_IN); // pooled+pkgsum contiguous
  float* pkg = pooled + N_TYPES * HC;

  for (int t = 0; t < N_TYPES; ++t) {
    const int* src = ei + (size_t)t * 2 * N_EDGES;
    const int* dst = src + N_EDGES;
    const float* xt  = x_tgt + (size_t)t * N_NODES * D_IN;
    const float* W1s = W1_src + (size_t)t * D_IN * HC;
    const float* W1d = W1_dst + (size_t)t * D_IN * HC;
    const float* W2s = W2_src + (size_t)t * D_IN * HC;
    const float* W2d = W2_dst + (size_t)t * HC * HC;
    const float* g_t = gamma + (size_t)t * HC;
    const float* be_t= beta  + (size_t)t * HC;

    // ---------- layer 1 ----------
    k_gemm_wmma<<<gGemm, 256, 0, stream>>>(x_pkg, W1s, bufA, N_NODES, D_IN);
    k_att<<<gNH, 256, 0, stream>>>(bufA, a1_src + (size_t)t * HC, es);
    k_gemm_wmma<<<gGemm, 256, 0, stream>>>(xt, W1d, bufB, N_NODES, D_IN);
    k_att<<<gNH, 256, 0, stream>>>(bufB, a1_dst + (size_t)t * HC, ed);
    k_init_sm<<<gNH, 256, 0, stream>>>(menc, den);
    k_edge_max<<<gE, 256, 0, stream>>>(src, dst, es, ed, menc);
    k_edge_expsum<<<gE, 256, 0, stream>>>(src, dst, es, ed, menc, den, exb);
    k_init_out<<<N_NODES, 256, 0, stream>>>(bufC, b1 + (size_t)t * HC);
    k_edge_agg<<<N_EDGES, 256, 0, stream>>>(src, dst, exb, den, bufA, bufC, nullptr);
    k_ln_relu<<<N_NODES, 256, 0, stream>>>(bufD, bufC, g_t, be_t);

    // ---------- layer 2 (src features are still raw x_pkg) ----------
    k_gemm_wmma<<<gGemm, 256, 0, stream>>>(x_pkg, W2s, bufA, N_NODES, D_IN);
    k_att<<<gNH, 256, 0, stream>>>(bufA, a2_src + (size_t)t * HC, es);
    k_gemm_wmma<<<gGemm, 256, 0, stream>>>(bufD, W2d, bufB, N_NODES, HC);
    k_att<<<gNH, 256, 0, stream>>>(bufB, a2_dst + (size_t)t * HC, ed);
    k_init_sm<<<gNH, 256, 0, stream>>>(menc, den);
    k_edge_max<<<gE, 256, 0, stream>>>(src, dst, es, ed, menc);
    k_edge_expsum<<<gE, 256, 0, stream>>>(src, dst, es, ed, menc, den, exb);
    k_init_out<<<N_NODES, 256, 0, stream>>>(bufC, b2 + (size_t)t * HC);
    float* alpha_out = out + 1 + (size_t)t * N_EDGES * H;
    k_edge_agg<<<N_EDGES, 256, 0, stream>>>(src, dst, exb, den, bufA, bufC, alpha_out);
    k_ln_relu<<<N_NODES, 256, 0, stream>>>(bufD, bufC, g_t, be_t);  // x2 overwrites x1

    // ---------- mean pool (sum; divide in final) ----------
    k_colsum<<<gRows, HC, 0, stream>>>(bufD, pooled + (size_t)t * HC, HC, N_NODES);
  }

  k_colsum<<<gRows, D_IN, 0, stream>>>(x_pkg, pkg, D_IN, N_NODES);
  k_final<<<1, 256, 0, stream>>>(pooled, pkg, Wc, bc, out);
}
